// MultiHeadAttention_30365418782948
// MI455X (gfx1250) — matrix operand order repaired and checked by reading, not compiled, not run on hardware
//
#include <hip/hip_runtime.h>

typedef __attribute__((ext_vector_type(16))) _Float16 v16h;
typedef __attribute__((ext_vector_type(8)))  _Float16 v8h;
typedef __attribute__((ext_vector_type(8)))  float    v8f;
typedef __attribute__((ext_vector_type(4)))  float    v4f;

typedef __attribute__((ext_vector_type(4))) unsigned int tdm_u32x4;
typedef __attribute__((ext_vector_type(8))) int          tdm_i32x8;
typedef __attribute__((ext_vector_type(4))) int          tdm_i32x4;

#define BB  4
#define SS  2048
#define HH  1024
#define NHH 16
#define HDD 64

// LDS row stride (floats) for the 16 x 2048 score panel: multiple of 4 for
// 16B ds_load_b128 alignment; 2052 % 64 == 4 so the 16 rows land on distinct
// banks when stage-3 A-fragments read one row per lane.
#define LSTRIDE 2052
#define ATTN_LDS_FLOATS (16 * LSTRIDE + 16 * 16 + 16 + 16 + 8 * 16 * 64)
#define ATTN_LDS_BYTES  (ATTN_LDS_FLOATS * 4 + 16 * HDD * 2)  // + f16 Q tile

// ---------------------------------------------------------------------------
// Fragment loaders (wave32, layouts per CDNA5 ISA 7.12.2)
// ---------------------------------------------------------------------------

// A-matrix 16x32 f16, row-major source, stride lda (elements).
__device__ __forceinline__ v16h load_a16(const _Float16* __restrict__ A, int lda,
                                         int m0, int k0, int lane) {
    const int row = m0 + (lane & 15);
    const int kb  = k0 + ((lane >> 4) << 3);
    const _Float16* p = A + (size_t)row * lda + kb;
    union { v16h f; v8h h[2]; } u;
    u.h[0] = *(const v8h*)(p);        // K = kb .. kb+7
    u.h[1] = *(const v8h*)(p + 16);   // K = kb+16 .. kb+23
    return u.f;
}

// B-matrix 32x16 f16 taken from a [N,K] row-major operand (X @ W^T form).
__device__ __forceinline__ v16h load_bT16(const _Float16* __restrict__ W, int ldw,
                                          int n0, int k0, int lane) {
    const int col = n0 + (lane & 15);
    const int kb  = k0 + ((lane >> 4) << 3);
    const _Float16* p = W + (size_t)col * ldw + kb;
    union { v16h f; v8h h[2]; } u;
    u.h[0] = *(const v8h*)(p);
    u.h[1] = *(const v8h*)(p + 16);
    return u.f;
}

// A-matrix 16x32 loaded from an fp32 source (global or LDS), cvt to f16.
__device__ __forceinline__ v16h load_a32(const float* __restrict__ A, int lda,
                                         int m0, int k0, int lane) {
    const int row = m0 + (lane & 15);
    const int kb  = k0 + ((lane >> 4) << 3);
    const float* p = A + (size_t)row * lda + kb;
    v4f f0 = *(const v4f*)(p);
    v4f f1 = *(const v4f*)(p + 4);
    v4f f2 = *(const v4f*)(p + 16);
    v4f f3 = *(const v4f*)(p + 20);
    v16h r;
#pragma unroll
    for (int i = 0; i < 4; ++i) {
        r[i]      = (_Float16)f0[i];
        r[4 + i]  = (_Float16)f1[i];
        r[8 + i]  = (_Float16)f2[i];
        r[12 + i] = (_Float16)f3[i];
    }
    return r;
}

// B-matrix 32x16 from an fp32 [N,K] row-major operand, cvt to f16.
__device__ __forceinline__ v16h load_bT32(const float* __restrict__ W, int ldw,
                                          int n0, int k0, int lane) {
    const int col = n0 + (lane & 15);
    const int kb  = k0 + ((lane >> 4) << 3);
    const float* p = W + (size_t)col * ldw + kb;
    v4f f0 = *(const v4f*)(p);
    v4f f1 = *(const v4f*)(p + 4);
    v4f f2 = *(const v4f*)(p + 16);
    v4f f3 = *(const v4f*)(p + 20);
    v16h r;
#pragma unroll
    for (int i = 0; i < 4; ++i) {
        r[i]      = (_Float16)f0[i];
        r[4 + i]  = (_Float16)f1[i];
        r[8 + i]  = (_Float16)f2[i];
        r[12 + i] = (_Float16)f3[i];
    }
    return r;
}

__device__ __forceinline__ v8f wmma32(v16h a, v16h b, v8f c) {
    return __builtin_amdgcn_wmma_f32_16x16x32_f16(false, a, false, b,
                                                  (short)0, c, false, false);
}

// ---------------------------------------------------------------------------
// Projection GEMM: out[m,n] = (sum_k X[m,k] * W[n,k] + bias[n]) * oscale
// fp32 inputs, f16 cvt in-register. MODE 0: out [B,NH,S,HD]; MODE 1: [B,NH,HD,S]
// oscale folds the attention 1/sqrt(HD) into the Q projection.
// ---------------------------------------------------------------------------
template <int MODE>
__global__ __launch_bounds__(256) void k_proj_gemm(const float* __restrict__ X,
                                                   const float* __restrict__ W,
                                                   const float* __restrict__ bias,
                                                   _Float16* __restrict__ out,
                                                   float oscale) {
    const int lane = threadIdx.x & 31;
    const int wave = blockIdx.x * (blockDim.x >> 5) + (threadIdx.x >> 5);
    const int NG = HH / 64;
    const int m0 = (wave / NG) * 16;
    const int n0 = (wave % NG) * 64;

    v8f acc[4] = {};
    for (int k0 = 0; k0 < HH; k0 += 32) {
        v16h a = load_a32(X, HH, m0, k0, lane);
#pragma unroll
        for (int j = 0; j < 4; ++j) {
            v16h b = load_bT32(W, HH, n0 + 16 * j, k0, lane);
            acc[j] = wmma32(a, b, acc[j]);
        }
    }

    const int mrow = m0 + ((lane >> 4) << 3);
    const int nc0  = n0 + (lane & 15);
#pragma unroll
    for (int j = 0; j < 4; ++j) {
        const int n = nc0 + 16 * j;
        const int h = n >> 6;
        const int d = n & 63;
        const float bv = bias[n];
#pragma unroll
        for (int v = 0; v < 8; ++v) {
            const int m  = mrow + v;
            const int b_ = m >> 11;     // / S
            const int s  = m & 2047;    // % S
            const float val = (acc[j][v] + bv) * oscale;
            size_t idx;
            if (MODE == 0) idx = ((size_t)(b_ * NHH + h) * SS + s) * HDD + d;
            else           idx = ((size_t)(b_ * NHH + h) * HDD + d) * SS + s;
            out[idx] = (_Float16)val;
        }
    }
}

// ---------------------------------------------------------------------------
// Fused attention: one workgroup (8 waves) per (b,h, 16-row q-tile).
//  Stage 0: TDM (tensor_load_to_lds) stages the 16x64 Q tile into LDS
//  Stage 1: scores (Q pre-scaled, causal tile-skip) -> LDS panel fp32
//  Stage 2: row softmax in LDS over valid prefix; suffix bulk-zeroed;
//           write normalized P to d_out (the only attn pass)
//  Stage 3: P@V from LDS, skipping fully-masked K-ranges
// All causal-skip decisions use SGPR-broadcast msk so EXEC stays all-1s
// around every WMMA (ISA 7.12 requirement).
// ---------------------------------------------------------------------------
__global__ __launch_bounds__(256) void k_attn_fused(const _Float16* __restrict__ Q,
                                                    const _Float16* __restrict__ Km,
                                                    const _Float16* __restrict__ Vt,
                                                    const int* __restrict__ causal,
                                                    float* __restrict__ attn,
                                                    _Float16* __restrict__ ctx) {
    extern __shared__ float lds[];
    float* sc     = lds;                       // [16][LSTRIDE] score panel
    float* red    = lds + 16 * LSTRIDE;        // [16][16] reduction scratch
    float* rowred = red + 16 * 16;             // [16] row max
    float* rowinv = rowred + 16;               // [16] 1/rowsum
    float* part   = rowinv + 16;               // [8][16][64] PV partials
    _Float16* qs  = (_Float16*)(lds + ATTN_LDS_FLOATS);  // [16][64] Q tile

    const int tid  = threadIdx.x;
    const int lane = tid & 31;
    const int wv   = tid >> 5;                 // wave 0..7

    const int bh = blockIdx.x >> 7;            // / (S/16)
    const int qt = blockIdx.x & 127;
    const int b_ = bh >> 4;
    const int h  = bh & 15;
    const int m0 = qt * 16;

    const _Float16* Qh = Q  + (size_t)bh * SS * HDD;
    const _Float16* Kh = Km + (size_t)bh * SS * HDD;
    const _Float16* Vh = Vt + (size_t)bh * HDD * SS;
    float* Ah = attn + (size_t)bh * SS * SS;

    // Prefetch the 256KB V panel for stage 3 (global_prefetch_b8 on gfx1250).
#pragma unroll
    for (int i = 0; i < 4; ++i)
        __builtin_prefetch((const void*)(Vh + ((size_t)tid * 4 + i) * 128), 0, 1);

    // ---- Stage 0: stage Q tile (16x64 f16, 2KB) into LDS ------------------
#if __has_builtin(__builtin_amdgcn_tensor_load_to_lds) && __has_builtin(__builtin_amdgcn_s_wait_tensorcnt)
    if (wv == 0) {                              // one TDM issue per workgroup
        const unsigned long long ga =
            (unsigned long long)(uintptr_t)(Qh + (size_t)m0 * HDD);
        const unsigned int ldsa = (unsigned int)(uintptr_t)qs; // LDS byte offset

        tdm_u32x4 g0;
        g0[0] = 1u;                             // count=1, user descriptor
        g0[1] = ldsa;                           // lds_addr
        g0[2] = (unsigned int)ga;               // global_addr[31:0]
        g0[3] = (unsigned int)(ga >> 32) | (2u << 30);  // addr[56:32] | type=2

        tdm_i32x8 g1;
        g1[0] = 1 << 16;                        // data_size = 2 bytes
        g1[1] = HDD << 16;                      // tensor_dim0 = 64
        g1[2] = SS << 16;                       // tensor_dim1 = 2048
        g1[3] = HDD << 16;                      // tile_dim0 = 64
        g1[4] = 16;                             // tile_dim1 = 16
        g1[5] = HDD;                            // tensor_dim0_stride = 64
        g1[6] = 0;
        g1[7] = 0;

        tdm_i32x4 gz = {0, 0, 0, 0};
#if defined(__clang_major__) && __clang_major__ >= 23
        tdm_i32x8 gz8 = {0, 0, 0, 0, 0, 0, 0, 0};
        __builtin_amdgcn_tensor_load_to_lds(g0, g1, gz, gz, gz8, 0);
#else
        __builtin_amdgcn_tensor_load_to_lds(g0, g1, gz, gz, 0);
#endif
        __builtin_amdgcn_s_wait_tensorcnt(0);
    }
#else
    // Fallback: cooperative copy of the Q tile.
    for (int i = tid; i < 16 * HDD / 8; i += 256)
        ((v8h*)qs)[i] = *(const v8h*)(Qh + (size_t)m0 * HDD + (size_t)i * 8);
#endif
    __syncthreads();

    // Causal geometry: broadcast to SGPR so skip branches are scalar
    // (s_cbranch_scc, EXEC untouched -> WMMA always sees EXEC all-1s).
    const int msk  = __builtin_amdgcn_readfirstlane(causal[0]);
    const int cend = msk ? ((m0 + 16 + 63) & ~63) : SS;  // 64-aligned valid end

    // ---- Stage 1: scores -> LDS (A-fragments from LDS Q tile) -------------
    const int  mrow = (lane >> 4) << 3;
    const int  ncol = lane & 15;
    const v16h a0 = load_a16(qs, HDD, 0, 0, lane);    // ds_load_b128
    const v16h a1 = load_a16(qs, HDD, 0, 32, lane);

#pragma unroll 4
    for (int t = 0; t < 16; ++t) {
        const int n0 = wv * 256 + t * 16;
        if (n0 >= cend) continue;               // region bulk-zeroed in stage 2

        const int c = n0 + ncol;
        if (msk && n0 > m0 + 15) {              // fully masked tile, no WMMA
#pragma unroll
            for (int v = 0; v < 8; ++v)
                sc[(mrow + v) * LSTRIDE + c] = -1.0e30f;
            continue;
        }

        v8f acc = {};
        acc = wmma32(a0, load_bT16(Kh, HDD, n0, 0, lane), acc);
        acc = wmma32(a1, load_bT16(Kh, HDD, n0, 32, lane), acc);

        if (msk && n0 + 15 > m0) {              // diagonal-straddling tile only
#pragma unroll
            for (int v = 0; v < 8; ++v) {
                float s = acc[v];
                if (c > m0 + mrow + v) s = -1.0e30f;
                sc[(mrow + v) * LSTRIDE + c] = s;
            }
        } else {
#pragma unroll
            for (int v = 0; v < 8; ++v)
                sc[(mrow + v) * LSTRIDE + c] = acc[v];
        }
    }
    __syncthreads();

    // ---- Stage 2: row softmax (vectorized, 16 threads x v4f per row) ------
    const int r  = tid >> 4;                    // row 0..15
    const int c4 = (tid & 15) * 4;              // starting float column

    float mx = -3.0e38f;
    for (int c = c4; c < cend; c += 64) {
        v4f x = *(const v4f*)(sc + r * LSTRIDE + c);
        mx = fmaxf(mx, fmaxf(fmaxf(x[0], x[1]), fmaxf(x[2], x[3])));
    }
    red[r * 16 + (tid & 15)] = mx;
    __syncthreads();
    if ((tid & 15) == 0) {
        float m2 = red[r * 16];
#pragma unroll
        for (int i = 1; i < 16; ++i) m2 = fmaxf(m2, red[r * 16 + i]);
        rowred[r] = m2;
    }
    __syncthreads();

    const float rmax = rowred[r];
    float sum = 0.0f;
    for (int c = c4; c < cend; c += 64) {
        v4f x = *(const v4f*)(sc + r * LSTRIDE + c);
        v4f e;
#pragma unroll
        for (int i = 0; i < 4; ++i) { e[i] = __expf(x[i] - rmax); sum += e[i]; }
        *(v4f*)(sc + r * LSTRIDE + c) = e;      // unnormalized; e <= 1
    }
    {   // bulk-zero the fully-masked suffix (exp of -inf)
        const v4f z = {0.0f, 0.0f, 0.0f, 0.0f};
        for (int c = cend + c4; c < SS; c += 64)
            *(v4f*)(sc + r * LSTRIDE + c) = z;
    }
    red[r * 16 + (tid & 15)] = sum;
    __syncthreads();
    if ((tid & 15) == 0) {
        float s2 = 0.0f;
#pragma unroll
        for (int i = 0; i < 16; ++i) s2 += red[r * 16 + i];
        rowinv[r] = 1.0f / s2;
    }
    __syncthreads();

    // normalized attention weights -> d_out, b128 stores (the 1GB stream)
    {
        const float inv = rowinv[r];
        float* Arow = Ah + (size_t)(m0 + r) * SS;
        for (int c = c4; c < SS; c += 64) {
            v4f x = *(const v4f*)(sc + r * LSTRIDE + c);
            v4f o;
#pragma unroll
            for (int i = 0; i < 4; ++i) o[i] = x[i] * inv;
            *(v4f*)(Arow + c) = o;
        }
    }

    // ---- Stage 3: P @ V from LDS, skipping fully-masked K-ranges ----------
    v8f cacc[4] = {};
    const int kbase = wv * 256;                 // each wave owns a K-range
    if (!(msk && kbase >= m0 + 16)) {           // scalar skip: P==0 there
#pragma unroll 2
        for (int ks = 0; ks < 8; ++ks) {
            const int k0 = kbase + ks * 32;
            v16h a = load_a32(sc, LSTRIDE, 0, k0, lane);   // ds_load + cvt
#pragma unroll
            for (int j = 0; j < 4; ++j) {
                v16h bfr = load_bT16(Vh, SS, 16 * j, k0, lane);
                cacc[j] = wmma32(a, bfr, cacc[j]);
            }
        }
    }
#pragma unroll
    for (int j = 0; j < 4; ++j) {
        const int d = ncol + 16 * j;
#pragma unroll
        for (int v = 0; v < 8; ++v)
            part[(wv * 16 + mrow + v) * 64 + d] = cacc[j][v];
    }
    __syncthreads();

    // cross-wave reduction + final 1/rowsum scaling -> ctx f16 [B,S,H]
    for (int e = tid; e < 16 * 64; e += 256) {
        const int m = e >> 6;
        const int d = e & 63;
        float s = 0.0f;
#pragma unroll
        for (int w2 = 0; w2 < 8; ++w2) s += part[(w2 * 16 + m) * 64 + d];
        s *= rowinv[m];
        ctx[(size_t)(b_ * SS + m0 + m) * HH + h * HDD + d] = (_Float16)s;
    }
}

// ---------------------------------------------------------------------------
// Output projection: out[m,n] = sum_k ctx[m,k] * Wo[n,k] + bo[n]  (fp32 out)
// ---------------------------------------------------------------------------
__global__ __launch_bounds__(256) void k_out_gemm(const _Float16* __restrict__ X,
                                                  const float* __restrict__ W,
                                                  const float* __restrict__ bias,
                                                  float* __restrict__ out) {
    const int lane = threadIdx.x & 31;
    const int wave = blockIdx.x * (blockDim.x >> 5) + (threadIdx.x >> 5);
    const int NG = HH / 64;
    const int m0 = (wave / NG) * 16;
    const int n0 = (wave % NG) * 64;

    v8f acc[4] = {};
    for (int k0 = 0; k0 < HH; k0 += 32) {
        v16h a = load_a16(X, HH, m0, k0, lane);
#pragma unroll
        for (int j = 0; j < 4; ++j) {
            v16h b = load_bT32(W, HH, n0 + 16 * j, k0, lane);
            acc[j] = wmma32(a, b, acc[j]);
        }
    }

    const int mrow = m0 + ((lane >> 4) << 3);
    const int nc0  = n0 + (lane & 15);
#pragma unroll
    for (int j = 0; j < 4; ++j) {
        const int n = nc0 + 16 * j;
        const float bv = bias[n];
#pragma unroll
        for (int v = 0; v < 8; ++v)
            out[(size_t)(mrow + v) * HH + n] = acc[j][v] + bv;
    }
}

// ---------------------------------------------------------------------------
extern "C" void kernel_launch(void* const* d_in, const int* in_sizes, int n_in,
                              void* d_out, int out_size, void* d_ws, size_t ws_size,
                              hipStream_t stream) {
    const float* query = (const float*)d_in[0];
    const float* key   = (const float*)d_in[1];
    const float* value = (const float*)d_in[2];
    const float* Wq    = (const float*)d_in[3];
    const float* bq    = (const float*)d_in[4];
    const float* Wk    = (const float*)d_in[5];
    const float* bk    = (const float*)d_in[6];
    const float* Wv    = (const float*)d_in[7];
    const float* bv    = (const float*)d_in[8];
    const float* Wo    = (const float*)d_in[9];
    const float* bo    = (const float*)d_in[10];
    const int*   causal = (const int*)d_in[11];

    float* out  = (float*)d_out;                       // [B,S,H] fp32
    float* attn = out + (size_t)BB * SS * HH;          // [B,NH,S,S] fp32

    const size_t perX = (size_t)BB * SS * HH;          // 8,388,608 elements

    _Float16* Qh  = (_Float16*)d_ws;                   // [B,NH,S,HD], pre-scaled
    _Float16* Kh  = Qh + perX;                         // [B,NH,S,HD]
    _Float16* Vt  = Kh + perX;                         // [B,NH,HD,S]
    _Float16* ctx = Vt + perX;                         // [B,S,H]

    // Q/K/V projections: 8192 waves -> 1024 blocks x 8 waves
    const int proj_blocks = (BB * SS / 16) * (HH / 64) / 8;
    k_proj_gemm<0><<<proj_blocks, 256, 0, stream>>>(query, Wq, bq, Qh, 0.125f);
    k_proj_gemm<0><<<proj_blocks, 256, 0, stream>>>(key,   Wk, bk, Kh, 1.0f);
    k_proj_gemm<1><<<proj_blocks, 256, 0, stream>>>(value, Wv, bv, Vt, 1.0f);

    // Fused scores + softmax + P@V: one block per (b,h,q-tile)
    const int attn_blocks = BB * NHH * (SS / 16);      // 8192
    k_attn_fused<<<attn_blocks, 256, ATTN_LDS_BYTES, stream>>>(Qh, Kh, Vt,
                                                               causal, attn, ctx);

    // Output projection (fp32 out)
    k_out_gemm<<<proj_blocks, 256, 0, stream>>>(ctx, Wo, bo, out);

    (void)in_sizes; (void)n_in; (void)out_size; (void)ws_size;
}
